// TransformerLayer_65541200937351
// MI455X (gfx1250) — compile-verified, matrix operand order checked by reading, not run on hardware
//
#include <hip/hip_runtime.h>
#include <hip/hip_bf16.h>
#include <cstdint>

// ---------------------------------------------------------------------------
// Problem constants (from reference): B=2, S=2048, D=2048, H=16, HD=128
// ---------------------------------------------------------------------------
#define BB 2
#define SS 2048
#define DD 2048
#define HH 16
#define HD_ 128
#define MM (BB * SS)   // 4096 token rows

typedef __bf16 bf16x16 __attribute__((ext_vector_type(16)));
typedef __bf16 bf16x4  __attribute__((ext_vector_type(4)));
typedef float  f32x8   __attribute__((ext_vector_type(8)));

// ---------------------------------------------------------------------------
// Tensor Data Mover (CDNA5): tensor_load_to_lds, TENSORcnt tracked.
// Descriptor packing per CDNA5 ISA.
// ---------------------------------------------------------------------------
#if defined(__has_builtin)
#  if __has_builtin(__builtin_amdgcn_tensor_load_to_lds) && \
      __has_builtin(__builtin_amdgcn_s_wait_tensorcnt)
#    define HAVE_TDM 1
#  endif
#endif
#ifndef HAVE_TDM
#  define HAVE_TDM 0
#endif

#if HAVE_TDM
typedef unsigned int u32x4 __attribute__((ext_vector_type(4)));
typedef int i32x4 __attribute__((ext_vector_type(4)));
typedef int i32x8 __attribute__((ext_vector_type(8)));

// 2D tile load: tile_h rows of tile_w bf16 elements, global row stride
// row_stride elems, into LDS at lds_addr, with LDS padding of pad_amount
// DWORD-code after every pad_interval DWORD-code (codes per D# spec).
__device__ __forceinline__ void tdm_load_2d_bf16(
    unsigned lds_addr, const void* gptr, unsigned tensor_w, unsigned tensor_h,
    unsigned tile_w, unsigned tile_h, unsigned row_stride,
    unsigned pad_interval_code, unsigned pad_amount_code) {
  unsigned long long ga = (unsigned long long)(uintptr_t)gptr;
  u32x4 g0;
  g0[0] = 1u;  // count=1, user descriptor, no gather
  g0[1] = lds_addr;
  g0[2] = (unsigned)(ga & 0xFFFFFFFFu);
  g0[3] = (unsigned)((ga >> 32) & 0x1FFFFFFu) | (2u << 30);  // type=2 (image)
  i32x8 g1;
  // data_size=1 (2 bytes) <<16 | pad_enable<<20 | pad_interval<<22 | pad_amt<<25
  g1[0] = (int)((1u << 16) | (1u << 20) | (pad_interval_code << 22) |
                (pad_amount_code << 25));
  g1[1] = (int)((tensor_w & 0xFFFFu) << 16);                       // dim0[15:0]
  g1[2] = (int)(((tensor_w >> 16) & 0xFFFFu) |
                ((tensor_h & 0xFFFFu) << 16));                     // dim0 hi, dim1 lo
  g1[3] = (int)(((tensor_h >> 16) & 0xFFFFu) |
                ((tile_w & 0xFFFFu) << 16));                       // dim1 hi, tile0
  g1[4] = (int)(tile_h & 0xFFFFu);                                 // tile1, tile2=0
  g1[5] = (int)row_stride;                                         // dim0_stride lo
  g1[6] = 0;                                                       // stride hi
  g1[7] = 0;
  i32x4 z4 = {0, 0, 0, 0};
#if __clang_major__ >= 23
  i32x8 z8 = {0, 0, 0, 0, 0, 0, 0, 0};
  __builtin_amdgcn_tensor_load_to_lds(g0, g1, z4, z4, z8, 0);
#else
  __builtin_amdgcn_tensor_load_to_lds(g0, g1, z4, z4, 0);
#endif
}
#endif  // HAVE_TDM

// ---------------------------------------------------------------------------
// f32 -> bf16 conversion (vectorized 4-wide)
// ---------------------------------------------------------------------------
__global__ void f32_to_bf16_kernel(const float* __restrict__ src,
                                   __bf16* __restrict__ dst, size_t n) {
  size_t i = ((size_t)blockIdx.x * blockDim.x + threadIdx.x) * 4;
  if (i + 3 < n) {
    float4 f = *(const float4*)(src + i);
    bf16x4 o = { (__bf16)f.x, (__bf16)f.y, (__bf16)f.z, (__bf16)f.w };
    *(bf16x4*)(dst + i) = o;
  }
}

// ---------------------------------------------------------------------------
// block-wide sum reduction (256 threads = 8 waves, wave32)
// ---------------------------------------------------------------------------
__device__ __forceinline__ float block_sum_256(float v) {
  __shared__ float red[8];
#pragma unroll
  for (int off = 16; off > 0; off >>= 1) v += __shfl_xor(v, off, 32);
  int w = threadIdx.x >> 5;
  if ((threadIdx.x & 31) == 0) red[w] = v;
  __syncthreads();
  float t = (threadIdx.x < 8) ? red[threadIdx.x] : 0.0f;
  if (threadIdx.x < 32) {
#pragma unroll
    for (int off = 4; off > 0; off >>= 1) t += __shfl_xor(t, off, 32);
    if (threadIdx.x == 0) red[0] = t;
  }
  __syncthreads();
  float r = red[0];
  __syncthreads();
  return r;
}

// ---------------------------------------------------------------------------
// LayerNorm over last dim (D=2048), one block per row, optional f32/bf16 out
// ---------------------------------------------------------------------------
__global__ __launch_bounds__(256) void layernorm_kernel(
    const float* __restrict__ x, const float* __restrict__ gamma,
    const float* __restrict__ beta, float* __restrict__ outF,
    __bf16* __restrict__ outB, int D) {
  size_t row = blockIdx.x;
  const float* xr = x + row * (size_t)D;
  float s = 0.f;
  for (int i = threadIdx.x; i < D; i += 256) s += xr[i];
  float mean = block_sum_256(s) / (float)D;
  float v = 0.f;
  for (int i = threadIdx.x; i < D; i += 256) {
    float d = xr[i] - mean;
    v += d * d;
  }
  float var = block_sum_256(v) / (float)D;
  float inv = rsqrtf(var + 1e-5f);
  for (int i = threadIdx.x; i < D; i += 256) {
    float y = (xr[i] - mean) * inv * gamma[i] + beta[i];
    if (outF) outF[row * (size_t)D + i] = y;
    if (outB) outB[row * (size_t)D + i] = (__bf16)y;
  }
}

// ---------------------------------------------------------------------------
// decb[d] = db[d] + sum_j pemb[trait, j] * dw[D + j, d]
// (persona half of concat([h,pers]) @ dw folds into a bias vector)
// ---------------------------------------------------------------------------
__global__ void decb_kernel(const float* __restrict__ pemb,
                            const int* __restrict__ trait,
                            const float* __restrict__ dw,
                            const float* __restrict__ db,
                            float* __restrict__ decb, int D) {
  int d = blockIdx.x * blockDim.x + threadIdx.x;
  if (d >= D) return;
  int t = trait[0];
  const float* pr = pemb + (size_t)t * D;
  float s = db[d];
  for (int j = 0; j < D; ++j) s += pr[j] * dw[(size_t)(D + j) * D + d];
  decb[d] = s;
}

// ---------------------------------------------------------------------------
// bf16 WMMA GEMM: C[M,N] = A[M,K] @ B[K,N]  (+bias)(+gelu)(+resid)
// Block tile 128x128, BK=32, 256 threads = 8 waves, each wave 32x64 (8 accs).
// A tile fed by the Tensor Data Mover (wave 0 issues, TENSORcnt wait),
// B tile transposed on LDS store. f32 acc via v_wmma_f32_16x16x32_bf16.
// ---------------------------------------------------------------------------
#define GBM 128
#define GBN 128
#define GBK 32
#define GPAD 8   // LDS row pad: 16B = 4 DWORDs -> matches TDM pad_amount code 3

template <bool GELU>
__global__ __launch_bounds__(256) void gemm_bf16_wmma(
    const __bf16* __restrict__ A, const __bf16* __restrict__ Bm,
    const float* __restrict__ bias, const float* __restrict__ resid,
    float* __restrict__ outF, __bf16* __restrict__ outB,
    int M, int N, int K) {
  __shared__ __bf16 sA[GBM][GBK + GPAD];
  __shared__ __bf16 sB[GBN][GBK + GPAD];  // stored transposed: [n][k]

  const int tid  = threadIdx.x;
  const int wid  = tid >> 5;          // 0..7
  const int lane = tid & 31;
  const int wm   = (wid & 3) * 32;    // wave M offset inside block tile
  const int wn   = (wid >> 2) * 64;   // wave N offset inside block tile
  const int m0   = blockIdx.y * GBM;
  const int n0   = blockIdx.x * GBN;
  const int lrow = lane & 15;
  const int ksel = (lane >> 4) << 4;  // 0 or 16

  f32x8 acc[2][4] = {};

  for (int k0 = 0; k0 < K; k0 += GBK) {
#if HAVE_TDM
    // ---- A tile via TDM: 128 x 32 bf16, row = 64B = 16 DWORDs (code 3),
    //      pad 16B = 4 DWORDs (code 3) to match the GPAD LDS layout.
    if (wid == 0) {
      tdm_load_2d_bf16((unsigned)(uintptr_t)&sA[0][0],
                       A + (size_t)m0 * K + k0, (unsigned)K, (unsigned)M,
                       GBK, GBM, (unsigned)K, 3u, 3u);
    }
#else
    {
      int r = tid >> 1, part = (tid & 1) << 4;
      const uint4* s = (const uint4*)(A + (size_t)(m0 + r) * K + k0 + part);
      uint4* d = (uint4*)&sA[r][part];
      d[0] = s[0];
      d[1] = s[1];
    }
#endif
    // ---- B tile: 32 k x 128 n, transposed into sB[n][k]
    {
      int k = tid >> 3, c0 = (tid & 7) << 4;
      union { uint4 u[2]; __bf16 h[16]; } tmp;
      const uint4* s = (const uint4*)(Bm + (size_t)(k0 + k) * N + n0 + c0);
      tmp.u[0] = s[0];
      tmp.u[1] = s[1];
#pragma unroll
      for (int i = 0; i < 16; ++i) sB[c0 + i][k] = tmp.h[i];
    }
    // prefetch next tiles while this one is consumed
    if (k0 + GBK < K) {
      __builtin_prefetch(Bm + (size_t)(k0 + GBK + (tid >> 3)) * N + n0 +
                             ((tid & 7) << 4), 0, 1);
    }
#if HAVE_TDM
    if (wid == 0) __builtin_amdgcn_s_wait_tensorcnt((short)0);
#endif
    __syncthreads();

    bf16x16 afrag[2], bfrag[4];
#pragma unroll
    for (int mi = 0; mi < 2; ++mi)
      afrag[mi] = *(const bf16x16*)&sA[wm + mi * 16 + lrow][ksel];
#pragma unroll
    for (int ni = 0; ni < 4; ++ni)
      bfrag[ni] = *(const bf16x16*)&sB[wn + ni * 16 + lrow][ksel];
#pragma unroll
    for (int mi = 0; mi < 2; ++mi)
#pragma unroll
      for (int ni = 0; ni < 4; ++ni)
        acc[mi][ni] = __builtin_amdgcn_wmma_f32_16x16x32_bf16(
            false, afrag[mi], false, bfrag[ni], (short)0, acc[mi][ni], false,
            false);
    __syncthreads();
  }

  // ---- epilogue: C tile layout: VGPR r -> row r (lanes 0-15) / row 8+r
  const int colb = n0 + wn + (lane & 15);
  const int rowb = m0 + wm + ((lane >> 4) ? 8 : 0);
#pragma unroll
  for (int mi = 0; mi < 2; ++mi) {
#pragma unroll
    for (int ni = 0; ni < 4; ++ni) {
      int col = colb + ni * 16;
      float bv = bias ? bias[col] : 0.0f;
#pragma unroll
      for (int r = 0; r < 8; ++r) {
        int row = rowb + mi * 16 + r;
        float v = acc[mi][ni][r] + bv;
        if (GELU) v = 0.5f * v * (1.0f + erff(v * 0.70710678118f));
        size_t idx = (size_t)row * N + col;
        if (resid) v += resid[idx];
        if (outF) outF[idx] = v;
        if (outB) outB[idx] = (__bf16)v;
      }
    }
  }
}

// ---------------------------------------------------------------------------
// Flash attention: grid (S/64, H, B), 128 threads = 4 waves.
// Q/K tiles via TDM; V transposed on LDS store; online softmax in f32;
// QK^T and P*V on WMMA. Each wave owns 16 query rows.
// ---------------------------------------------------------------------------
__global__ __launch_bounds__(128) void flash_attn_kernel(
    const __bf16* __restrict__ Q, const __bf16* __restrict__ Kb,
    const __bf16* __restrict__ V, const int* __restrict__ amask,
    __bf16* __restrict__ O, float scale) {
  __shared__ __bf16 sQ[64][HD_ + 8];   // [q][hd]    row 256B + 16B pad
  __shared__ __bf16 sK[64][HD_ + 8];   // [key][hd]
  __shared__ __bf16 sVt[HD_][64 + 8];  // transposed: [hd][key]
  __shared__ __bf16 sP[4][16][64];     // per-wave P tile (A-frag staging)

  const int tid  = threadIdx.x;
  const int wid  = tid >> 5;
  const int lane = tid & 31;
  const int lrow = lane & 15;
  const int ksel = (lane >> 4) << 4;
  const int b    = blockIdx.z;
  const int h    = blockIdx.y;
  const int q0   = blockIdx.x * 64;
  const int hcol = h * HD_;
  const size_t qbase = (size_t)(b * SS + q0) * DD;

  // load Q tile (64 x 128): row = 256B = 64 DWORDs (code 5), pad 4 DW (code 3)
#if HAVE_TDM
  if (wid == 0) {
    tdm_load_2d_bf16((unsigned)(uintptr_t)&sQ[0][0], Q + qbase + hcol,
                     (unsigned)DD, 64u, HD_, 64u, (unsigned)DD, 5u, 3u);
    __builtin_amdgcn_s_wait_tensorcnt((short)0);
  }
#else
  for (int c = tid; c < 512; c += 128) {
    int r = c >> 3, p = (c & 7) << 4;
    const uint4* s = (const uint4*)(Q + qbase + (size_t)r * DD + hcol + p);
    uint4* d = (uint4*)&sQ[r][p];
    d[0] = s[0];
    d[1] = s[1];
  }
#endif

  float mrun[8], lrun[8];
  f32x8 accO[8] = {};
#pragma unroll
  for (int r = 0; r < 8; ++r) {
    mrun[r] = -3.0e38f;
    lrun[r] = 0.0f;
  }
  __syncthreads();

  for (int kb = 0; kb < SS; kb += 64) {
    const size_t kbase = (size_t)(b * SS + kb) * DD;
    // K tile
#if HAVE_TDM
    if (wid == 0) {
      tdm_load_2d_bf16((unsigned)(uintptr_t)&sK[0][0], Kb + kbase + hcol,
                       (unsigned)DD, 64u, HD_, 64u, (unsigned)DD, 5u, 3u);
    }
#else
    for (int c = tid; c < 512; c += 128) {
      int r = c >> 3, p = (c & 7) << 4;
      const uint4* s = (const uint4*)(Kb + kbase + (size_t)r * DD + hcol + p);
      uint4* d = (uint4*)&sK[r][p];
      d[0] = s[0];
      d[1] = s[1];
    }
#endif
    // V tile, transposed into sVt[hd][key]
    for (int c = tid; c < 512; c += 128) {
      int r = c >> 3, p = (c & 7) << 4;
      union { uint4 u[2]; __bf16 hv[16]; } tmp;
      const uint4* s = (const uint4*)(V + kbase + (size_t)r * DD + hcol + p);
      tmp.u[0] = s[0];
      tmp.u[1] = s[1];
#pragma unroll
      for (int i = 0; i < 16; ++i) sVt[p + i][r] = tmp.hv[i];
    }
#if HAVE_TDM
    if (wid == 0) __builtin_amdgcn_s_wait_tensorcnt((short)0);
#endif
    __syncthreads();

    // ---- S = scale * Q K^T  (16 q rows x 64 keys per wave)
    f32x8 sacc[4] = {};
#pragma unroll
    for (int ni = 0; ni < 4; ++ni) {
#pragma unroll
      for (int kk = 0; kk < 4; ++kk) {
        bf16x16 a = *(const bf16x16*)&sQ[wid * 16 + lrow][kk * 32 + ksel];
        bf16x16 bf = *(const bf16x16*)&sK[ni * 16 + lrow][kk * 32 + ksel];
        sacc[ni] = __builtin_amdgcn_wmma_f32_16x16x32_bf16(
            false, a, false, bf, (short)0, sacc[ni], false, false);
      }
    }

    // ---- mask + online softmax (rows live in one 16-lane half)
    float cmax[8];
#pragma unroll
    for (int r = 0; r < 8; ++r) cmax[r] = -3.0e38f;
#pragma unroll
    for (int ni = 0; ni < 4; ++ni) {
      int key = kb + ni * 16 + lrow;
      float madd = amask[(size_t)b * SS + key] ? 0.0f : -1.0e30f;
#pragma unroll
      for (int r = 0; r < 8; ++r) {
        float sv = sacc[ni][r] * scale + madd;
        sacc[ni][r] = sv;
        cmax[r] = fmaxf(cmax[r], sv);
      }
    }
#pragma unroll
    for (int off = 8; off > 0; off >>= 1)
#pragma unroll
      for (int r = 0; r < 8; ++r)
        cmax[r] = fmaxf(cmax[r], __shfl_xor(cmax[r], off, 32));

    float alpha[8], rsum[8];
#pragma unroll
    for (int r = 0; r < 8; ++r) {
      float mn = fmaxf(mrun[r], cmax[r]);
      alpha[r] = __expf(mrun[r] - mn);
      mrun[r] = mn;
      rsum[r] = 0.0f;
    }
#pragma unroll
    for (int ni = 0; ni < 4; ++ni)
#pragma unroll
      for (int r = 0; r < 8; ++r) {
        float p = __expf(sacc[ni][r] - mrun[r]);
        sacc[ni][r] = p;
        rsum[r] += p;
      }
#pragma unroll
    for (int off = 8; off > 0; off >>= 1)
#pragma unroll
      for (int r = 0; r < 8; ++r) rsum[r] += __shfl_xor(rsum[r], off, 32);
#pragma unroll
    for (int r = 0; r < 8; ++r) lrun[r] = lrun[r] * alpha[r] + rsum[r];
#pragma unroll
    for (int t = 0; t < 8; ++t)
#pragma unroll
      for (int r = 0; r < 8; ++r) accO[t][r] *= alpha[r];

    // ---- stage P (C layout) -> LDS -> reload as A fragments (same-wave DS
    //      accesses stay in order per the DS pipeline)
    {
      int prow = (lane >= 16) ? 8 : 0;
#pragma unroll
      for (int ni = 0; ni < 4; ++ni)
#pragma unroll
        for (int r = 0; r < 8; ++r)
          sP[wid][prow + r][ni * 16 + lrow] = (__bf16)sacc[ni][r];
    }

    // ---- O += P @ V  (K = 64 keys -> 2 wmma steps; 8 hd tiles)
#pragma unroll
    for (int t = 0; t < 8; ++t) {
#pragma unroll
      for (int ks = 0; ks < 2; ++ks) {
        bf16x16 a = *(const bf16x16*)&sP[wid][lrow][ks * 32 + ksel];
        bf16x16 bf = *(const bf16x16*)&sVt[t * 16 + lrow][ks * 32 + ksel];
        accO[t] = __builtin_amdgcn_wmma_f32_16x16x32_bf16(
            false, a, false, bf, (short)0, accO[t], false, false);
      }
    }
    __syncthreads();  // protect sK/sVt before next tile load
  }

  // ---- finalize: O /= l, write bf16
  float inv[8];
#pragma unroll
  for (int r = 0; r < 8; ++r) inv[r] = 1.0f / lrun[r];
  const int rowb = q0 + wid * 16 + ((lane >> 4) ? 8 : 0);
#pragma unroll
  for (int t = 0; t < 8; ++t) {
    int col = hcol + t * 16 + lrow;
#pragma unroll
    for (int r = 0; r < 8; ++r) {
      int row = rowb + r;
      O[(size_t)(b * SS + row) * DD + col] = (__bf16)(accO[t][r] * inv[r]);
    }
  }
}

// ---------------------------------------------------------------------------
// Orchestration
// ---------------------------------------------------------------------------
extern "C" void kernel_launch(void* const* d_in, const int* in_sizes, int n_in,
                              void* d_out, int out_size, void* d_ws,
                              size_t ws_size, hipStream_t stream) {
  const float* hs    = (const float*)d_in[0];
  const int*   amask = (const int*)d_in[1];
  const int*   trait = (const int*)d_in[2];
  const float* wq = (const float*)d_in[3];
  const float* wk = (const float*)d_in[4];
  const float* wv = (const float*)d_in[5];
  const float* wo = (const float*)d_in[6];
  const float* g1 = (const float*)d_in[7];
  const float* b1 = (const float*)d_in[8];
  const float* g2 = (const float*)d_in[9];
  const float* b2 = (const float*)d_in[10];
  const float* rw1 = (const float*)d_in[11];
  const float* rb1 = (const float*)d_in[12];
  const float* rw2 = (const float*)d_in[13];
  const float* rb2 = (const float*)d_in[14];
  const float* rg  = (const float*)d_in[15];
  const float* rbt = (const float*)d_in[16];
  const float* pemb = (const float*)d_in[17];
  const float* dwi  = (const float*)d_in[18];
  const float* db   = (const float*)d_in[19];
  const float* wg   = (const float*)d_in[20];
  const float* wb   = (const float*)d_in[21];
  const float* mw1  = (const float*)d_in[22];
  const float* mb1  = (const float*)d_in[23];
  const float* mw2  = (const float*)d_in[24];
  const float* mb2  = (const float*)d_in[25];

  // workspace bump allocator (256B aligned)
  size_t off = 0;
  char* base = (char*)d_ws;
  auto take = [&](size_t bytes) -> void* {
    void* p = base + off;
    off += (bytes + 255) & ~(size_t)255;
    return p;
  };

  __bf16* wq_b  = (__bf16*)take((size_t)DD * DD * 2);
  __bf16* wk_b  = (__bf16*)take((size_t)DD * DD * 2);
  __bf16* wv_b  = (__bf16*)take((size_t)DD * DD * 2);
  __bf16* wo_b  = (__bf16*)take((size_t)DD * DD * 2);
  __bf16* rw1_b = (__bf16*)take((size_t)DD * 2 * DD * 2);
  __bf16* rw2_b = (__bf16*)take((size_t)2 * DD * DD * 2);
  __bf16* dwt_b = (__bf16*)take((size_t)DD * DD * 2);
  __bf16* mw1_b = (__bf16*)take((size_t)DD * 4 * DD * 2);
  __bf16* mw2_b = (__bf16*)take((size_t)4 * DD * DD * 2);

  float* x2    = (float*)take((size_t)MM * DD * 4);
  float* h2f   = (float*)take((size_t)MM * DD * 4);
  float* hr    = (float*)take((size_t)MM * DD * 4);   // reused as h4pre
  float* h3f   = (float*)take((size_t)MM * DD * 4);
  float* decb  = (float*)take((size_t)DD * 4);

  __bf16* h1bf   = (__bf16*)take((size_t)MM * DD * 2);
  __bf16* qbf    = (__bf16*)take((size_t)MM * DD * 2);
  __bf16* kbf    = (__bf16*)take((size_t)MM * DD * 2);
  __bf16* vbf    = (__bf16*)take((size_t)MM * DD * 2);
  __bf16* attnbf = (__bf16*)take((size_t)MM * DD * 2);
  __bf16* h2bf   = (__bf16*)take((size_t)MM * DD * 2);
  __bf16* r1bf   = (__bf16*)take((size_t)MM * 2 * DD * 2);
  __bf16* h3bf   = (__bf16*)take((size_t)MM * DD * 2);
  __bf16* h4bf   = (__bf16*)take((size_t)MM * DD * 2);
  __bf16* m1bf   = (__bf16*)take((size_t)MM * 4 * DD * 2);
  (void)ws_size;
  (void)in_sizes; (void)n_in; (void)out_size;

  auto conv = [&](const float* s, __bf16* d, size_t n) {
    unsigned blocks = (unsigned)((n / 4 + 255) / 256);
    f32_to_bf16_kernel<<<blocks, 256, 0, stream>>>(s, d, n);
  };
  auto gemm = [&](const __bf16* A, const __bf16* Bw, const float* bias,
                  const float* resid, float* oF, __bf16* oB, int M_, int N_,
                  int K_, bool gelu) {
    dim3 g(N_ / GBN, M_ / GBM);
    if (gelu)
      gemm_bf16_wmma<true><<<g, 256, 0, stream>>>(A, Bw, bias, resid, oF, oB,
                                                  M_, N_, K_);
    else
      gemm_bf16_wmma<false><<<g, 256, 0, stream>>>(A, Bw, bias, resid, oF, oB,
                                                   M_, N_, K_);
  };

  // 1. weight conversions to bf16 (dw: only top DxD block feeds the GEMM)
  conv(wq, wq_b, (size_t)DD * DD);
  conv(wk, wk_b, (size_t)DD * DD);
  conv(wv, wv_b, (size_t)DD * DD);
  conv(wo, wo_b, (size_t)DD * DD);
  conv(rw1, rw1_b, (size_t)DD * 2 * DD);
  conv(rw2, rw2_b, (size_t)2 * DD * DD);
  conv(dwi, dwt_b, (size_t)DD * DD);
  conv(mw1, mw1_b, (size_t)DD * 4 * DD);
  conv(mw2, mw2_b, (size_t)4 * DD * DD);

  // 2. h1 = LN(x, g1, b1) -> bf16
  layernorm_kernel<<<MM, 256, 0, stream>>>(hs, g1, b1, nullptr, h1bf, DD);

  // 3. q/k/v projections
  gemm(h1bf, wq_b, nullptr, nullptr, nullptr, qbf, MM, DD, DD, false);
  gemm(h1bf, wk_b, nullptr, nullptr, nullptr, kbf, MM, DD, DD, false);
  gemm(h1bf, wv_b, nullptr, nullptr, nullptr, vbf, MM, DD, DD, false);

  // 4. flash attention
  flash_attn_kernel<<<dim3(SS / 64, HH, BB), 128, 0, stream>>>(
      qbf, kbf, vbf, amask, attnbf, 0.08838834764831845f /* 1/sqrt(128) */);

  // 5. x2 = hidden + attn @ wo
  gemm(attnbf, wo_b, nullptr, hs, x2, nullptr, MM, DD, DD, false);

  // 6. h2 = LN(x2, g2, b2)
  layernorm_kernel<<<MM, 256, 0, stream>>>(x2, g2, b2, h2f, h2bf, DD);

  // 7. r1 = gelu(h2 @ rw1 + rb1)
  gemm(h2bf, rw1_b, rb1, nullptr, nullptr, r1bf, MM, 2 * DD, DD, true);

  // 8. hr = h2 + (r1 @ rw2 + rb2)
  gemm(r1bf, rw2_b, rb2, h2f, hr, nullptr, MM, DD, 2 * DD, false);

  // 9. h3 = LN(hr, rg, rb)
  layernorm_kernel<<<MM, 256, 0, stream>>>(hr, rg, rbt, h3f, h3bf, DD);

  // 10. decb = db + pemb[trait] @ dw_bottom
  decb_kernel<<<DD / 256, 256, 0, stream>>>(pemb, trait, dwi, db, decb, DD);

  // 11. h4pre = h3 + (h3 @ dw_top + decb)   (reuse hr)
  gemm(h3bf, dwt_b, decb, h3f, hr, nullptr, MM, DD, DD, false);

  // 12. h4 = LN(h4pre, wg, wb)
  layernorm_kernel<<<MM, 256, 0, stream>>>(hr, wg, wb, nullptr, h4bf, DD);

  // 13. m1 = gelu(h4 @ mw1 + mb1)
  gemm(h4bf, mw1_b, mb1, nullptr, nullptr, m1bf, MM, 4 * DD, DD, true);

  // 14. out = x2 + (m1 @ mw2 + mb2)
  gemm(m1bf, mw2_b, mb2, x2, (float*)d_out, nullptr, MM, DD, 4 * DD, false);
}
